// SimpleMPNN_15814069584048
// MI455X (gfx1250) — matrix operand order repaired and checked by reading, not compiled, not run on hardware
//
#include <hip/hip_runtime.h>
#include <hip/hip_bf16.h>

// ---------------------------------------------------------------------------
// Types for CDNA5 WMMA
// ---------------------------------------------------------------------------
typedef __bf16 v16bf __attribute__((ext_vector_type(16)));
typedef float  v8f   __attribute__((ext_vector_type(8)));

struct U8 { unsigned u[8]; };

__device__ __forceinline__ v16bf frag_cast(const U8& t) {
    return __builtin_bit_cast(v16bf, t);
}

__device__ __forceinline__ v8f wmma_bf16(v16bf a, v16bf b, v8f c) {
    // D = A(16x32 bf16) * B(32x16 bf16) + C(16x16 f32)
    return __builtin_amdgcn_wmma_f32_16x16x32_bf16(
        /*neg_a=*/false, a, /*neg_b=*/false, b,
        /*c_mod=*/(short)0, c, /*reuse_a=*/false, /*reuse_b=*/false);
}

__device__ __forceinline__ unsigned short f2bf(float f) {
    unsigned u = __builtin_bit_cast(unsigned, f);
    return (unsigned short)((u + 0x7FFFu + ((u >> 16) & 1u)) >> 16);
}
__device__ __forceinline__ unsigned pack2(unsigned short lo, unsigned short hi) {
    return (unsigned)lo | ((unsigned)hi << 16);
}
__device__ __forceinline__ float sigmoidf_fast(float x) {
    return 1.0f / (1.0f + __expf(-x));
}
__device__ __forceinline__ float tanhf_fast(float x) {
    float t = __expf(2.0f * x);
    return (t - 1.0f) / (t + 1.0f);
}

// gfx1250 async global->LDS DMA (ASYNCcnt-tracked), 16B per lane
__device__ __forceinline__ void async_load_b128(unsigned lds_byte_off,
                                                unsigned long long gaddr) {
    asm volatile("global_load_async_to_lds_b128 %0, %1, off"
                 :: "v"(lds_byte_off), "v"(gaddr) : "memory");
}
__device__ __forceinline__ void wait_async0() {
    asm volatile("s_wait_asynccnt 0x0" ::: "memory");
}

#define Hdim 512
#define ROWU 20  // LDS tile row stride in dwords (80B: b128-aligned, bank-safe)

// ---------------------------------------------------------------------------
// Weight conversion: f32 [L][K][Nout] -> pair-packed column-major bf16
// Wt[l][n][p] (uint) = { bf16(W[2p][n]), bf16(W[2p+1][n]) }  — this is exactly
// the LDS B-tile layout, so B staging becomes a raw async b128 DMA.
// ---------------------------------------------------------------------------
__global__ void convert_transpose_kernel(const float* __restrict__ W,
                                         unsigned* __restrict__ Wt,
                                         int K, int Nout, long total) {
    long i = (long)blockIdx.x * blockDim.x + threadIdx.x;
    long stride = (long)gridDim.x * blockDim.x;
    const int K2 = K >> 1;
    for (; i < total; i += stride) {
        int n = (int)(i % Nout);
        long rest = i / Nout;
        int p = (int)(rest % K2);
        int l = (int)(rest / K2);
        const float* Wl = W + (size_t)l * K * Nout;
        unsigned short lo = f2bf(Wl[(size_t)(2 * p) * Nout + n]);
        unsigned short hi = f2bf(Wl[(size_t)(2 * p + 1) * Nout + n]);
        Wt[((size_t)l * Nout + n) * K2 + p] = pack2(lo, hi);
    }
}

// ---------------------------------------------------------------------------
// Encoder: h[i][j] = x[i][:24] @ enc_w[:, j] + enc_b[j]; also writes bf16 copy
// ---------------------------------------------------------------------------
__global__ __launch_bounds__(256) void encoder_kernel(
    const float* __restrict__ x, const float* __restrict__ w,
    const float* __restrict__ b, float* __restrict__ h,
    unsigned short* __restrict__ hbf, int N) {
    int i = blockIdx.x;
    if (i >= N) return;
    __shared__ float xs[24];
    if (threadIdx.x < 24) xs[threadIdx.x] = x[(size_t)i * 24 + threadIdx.x];
    __syncthreads();
    for (int j = threadIdx.x; j < Hdim; j += 256) {
        float acc = b[j];
        #pragma unroll
        for (int k = 0; k < 24; k++) acc = fmaf(xs[k], w[k * Hdim + j], acc);
        h[(size_t)i * Hdim + j] = acc;
        hbf[(size_t)i * Hdim + j] = f2bf(acc);
    }
}

// ---------------------------------------------------------------------------
// bf16 WMMA GEMM: out = (relu?)(A[M,K]bf16 @ B[K,Nout]bf16 + bias)
// A and B tiles both fetched via async global->LDS DMA, double-buffered.
// Block tile 128x64, 8 waves, each wave a 32x32 region (2x2 WMMA tiles).
// mode 0: f32 output to Cf;  mode 1: relu + bf16 output to Cb.
// ---------------------------------------------------------------------------
__global__ __launch_bounds__(256) void gemm_bias_kernel(
    const unsigned short* __restrict__ A, const unsigned* __restrict__ Bt,
    const float* __restrict__ bias, float* __restrict__ Cf,
    unsigned short* __restrict__ Cb, int M, int Nout, int K, int mode) {
    __shared__ unsigned ldsA[2 * 128 * ROWU];
    __shared__ unsigned ldsB[2 * 64 * ROWU];

    const int tid  = threadIdx.x;
    const int lane = tid & 31;
    const int wave = tid >> 5;
    const int wm = (wave >> 1) * 32;
    const int wn = (wave & 1) * 32;
    const int row0 = blockIdx.x * 128;
    const int col0 = blockIdx.y * 64;
    const unsigned ldsA_base = (unsigned)(size_t)&ldsA[0];
    const unsigned ldsB_base = (unsigned)(size_t)&ldsB[0];
    const int K2 = K >> 1;
    const int nsteps = K / 32;

    v8f acc[2][2];
    #pragma unroll
    for (int i = 0; i < 2; i++)
        #pragma unroll
        for (int j = 0; j < 2; j++) acc[i][j] = (v8f){0.f,0.f,0.f,0.f,0.f,0.f,0.f,0.f};

    auto issue_asyncA = [&](int buf, int k0) {
        #pragma unroll
        for (int q = 0; q < 2; q++) {
            int id = wave * 64 + q * 32 + lane;   // 512 chunks of 16B
            int r = id >> 2, c = id & 3;
            int rowg = row0 + r;
            if (rowg > M - 1) rowg = M - 1;       // clamp tail (masked at store)
            unsigned long long ga =
                (unsigned long long)(const void*)(A + (size_t)rowg * K + k0) +
                (unsigned)(c * 16);
            unsigned la = ldsA_base + (unsigned)(buf * 128 * 80 + r * 80 + c * 16);
            async_load_b128(la, ga);
        }
    };
    auto issue_asyncB = [&](int buf, int k0) {
        int n = tid >> 2, c = tid & 3;            // 256 chunks of 16B
        unsigned long long ga =
            (unsigned long long)(const void*)(Bt + (size_t)(col0 + n) * K2 + (k0 >> 1)) +
            (unsigned)(c * 16);
        unsigned la = ldsB_base + (unsigned)(buf * 64 * 80 + n * 80 + c * 16);
        async_load_b128(la, ga);
    };

    issue_asyncA(0, 0);
    issue_asyncB(0, 0);

    const int lm  = lane & 15;
    const int kb2 = (lane >> 4) * 4;
    const int bb  = (lane >> 4) * 4;

    for (int s = 0; s < nsteps; s++) {
        wait_async0();
        __syncthreads();
        if (s + 1 < nsteps) {
            issue_asyncA((s + 1) & 1, 32 * (s + 1));
            issue_asyncB((s + 1) & 1, 32 * (s + 1));
        }
        const int cur = s & 1;
        const unsigned* __restrict__ Abuf = &ldsA[cur * 128 * ROWU];
        const unsigned* __restrict__ Bbuf = &ldsB[cur * 64 * ROWU];

        v16bf afrag[2], bfrag[2];
        #pragma unroll
        for (int i = 0; i < 2; i++) {
            U8 t;
            int base = (wm + i * 16 + lm) * ROWU;
            #pragma unroll
            for (int p = 0; p < 4; p++) {
                t.u[p]     = Abuf[base + kb2 + p];
                t.u[4 + p] = Abuf[base + 8 + kb2 + p];
            }
            afrag[i] = frag_cast(t);
        }
        #pragma unroll
        for (int j = 0; j < 2; j++) {
            U8 t;
            int base = (wn + j * 16 + lm) * ROWU + bb;
            #pragma unroll
            for (int p = 0; p < 8; p++) t.u[p] = Bbuf[base + (p & 3) + (p >> 2) * 8];
            bfrag[j] = frag_cast(t);
        }
        #pragma unroll
        for (int i = 0; i < 2; i++)
            #pragma unroll
            for (int j = 0; j < 2; j++)
                acc[i][j] = wmma_bf16(afrag[i], bfrag[j], acc[i][j]);
    }

    // ---- epilogue (mode hoisted out of the store loops) ----
    const int mhalf = (lane >> 4) * 8;
    if (mode == 1) {
        #pragma unroll
        for (int i = 0; i < 2; i++)
            #pragma unroll
            for (int j = 0; j < 2; j++) {
                int col = col0 + wn + j * 16 + lm;
                float bv = bias[col];
                #pragma unroll
                for (int v = 0; v < 8; v++) {
                    int row = row0 + wm + i * 16 + mhalf + v;
                    if (row < M)
                        Cb[(size_t)row * Nout + col] = f2bf(fmaxf(acc[i][j][v] + bv, 0.f));
                }
            }
    } else {
        #pragma unroll
        for (int i = 0; i < 2; i++)
            #pragma unroll
            for (int j = 0; j < 2; j++) {
                int col = col0 + wn + j * 16 + lm;
                float bv = bias[col];
                #pragma unroll
                for (int v = 0; v < 8; v++) {
                    int row = row0 + wm + i * 16 + mhalf + v;
                    if (row < M)
                        Cf[(size_t)row * Nout + col] = acc[i][j][v] + bv;
                }
            }
    }
}

// ---------------------------------------------------------------------------
// m = msg (self-loop init), then scatter-add over edges
// ---------------------------------------------------------------------------
__global__ void copy4_kernel(const float4* __restrict__ s, float4* __restrict__ d, long n) {
    long i = (long)blockIdx.x * blockDim.x + threadIdx.x;
    long stride = (long)gridDim.x * blockDim.x;
    for (; i < n; i += stride) d[i] = s[i];
}

__global__ __launch_bounds__(128) void scatter_kernel(
    const int* __restrict__ edge, const float* __restrict__ msg,
    float* __restrict__ m, int E) {
    int e = blockIdx.x;
    if (e >= E) return;
    int src = edge[e];
    int dst = edge[E + e];
    const float4* ms = (const float4*)(msg + (size_t)src * Hdim);
    float* md = m + (size_t)dst * Hdim;
    int t = threadIdx.x;
    float4 v = ms[t];
    atomicAdd(md + 4 * t + 0, v.x);
    atomicAdd(md + 4 * t + 1, v.y);
    atomicAdd(md + 4 * t + 2, v.z);
    atomicAdd(md + 4 * t + 3, v.w);
}

// ---------------------------------------------------------------------------
// Fused GRU: per block, 128x16 tile of h_next; all 6 gate GEMM tiles in-reg.
// Double-buffered: h operand async-DMA'd from bf16 shadow, 6 B tiles async-
// DMA'd from transposed-packed weights; only m (f32 from scatter) converts.
// ---------------------------------------------------------------------------
__global__ __launch_bounds__(256) void gru_kernel(
    const float* __restrict__ Mm, const float* __restrict__ Hm,
    const unsigned short* __restrict__ Hbfin,
    const unsigned* __restrict__ Wiht, const unsigned* __restrict__ Whht,
    const float* __restrict__ bih, const float* __restrict__ bhh,
    float* __restrict__ Hout, unsigned short* __restrict__ Hbf, int Mrows) {
    __shared__ unsigned ldsM[2][128][17];         // m tile (staged + converted)
    __shared__ unsigned ldsH[2 * 128 * ROWU];     // h tile (async bf16 DMA)
    __shared__ unsigned ldsB[2 * 6 * 16 * ROWU];  // 6 gate B tiles (async DMA)

    const int tid  = threadIdx.x;
    const int lane = tid & 31;
    const int wave = tid >> 5;
    const int row0 = blockIdx.x * 128;
    const int col0 = blockIdx.y * 16;
    const unsigned ldsH_base = (unsigned)(size_t)&ldsH[0];
    const unsigned ldsB_base = (unsigned)(size_t)&ldsB[0];
    const int K2 = Hdim >> 1;  // 256 uints per weight column
    const int nsteps = Hdim / 32;

    v8f aI[3], aH[3];
    #pragma unroll
    for (int g = 0; g < 3; g++) {
        aI[g] = (v8f){0.f,0.f,0.f,0.f,0.f,0.f,0.f,0.f};
        aH[g] = (v8f){0.f,0.f,0.f,0.f,0.f,0.f,0.f,0.f};
    }

    auto issue_asyncH = [&](int buf, int k0) {
        #pragma unroll
        for (int q = 0; q < 2; q++) {
            int id = wave * 64 + q * 32 + lane;   // 512 chunks of 16B
            int r = id >> 2, c = id & 3;
            int rowg = row0 + r;
            if (rowg > Mrows - 1) rowg = Mrows - 1;
            unsigned long long ga =
                (unsigned long long)(const void*)(Hbfin + (size_t)rowg * Hdim + k0) +
                (unsigned)(c * 16);
            unsigned la = ldsH_base + (unsigned)(buf * 128 * 80 + r * 80 + c * 16);
            async_load_b128(la, ga);
        }
    };
    auto issue_asyncB = [&](int buf, int k0) {
        #pragma unroll
        for (int q = 0; q < 2; q++) {
            int id = tid + 256 * q;               // 384 chunks of 16B
            if (id < 384) {
                int g = id >> 6, rem = id & 63;
                int n = rem >> 2, c = rem & 3;
                const unsigned* W = (g < 3) ? Wiht : Whht;
                int gg = (g < 3) ? g : g - 3;
                int col = gg * Hdim + col0 + n;
                unsigned long long ga =
                    (unsigned long long)(const void*)(W + (size_t)col * K2 + (k0 >> 1)) +
                    (unsigned)(c * 16);
                unsigned la = ldsB_base +
                              (unsigned)(buf * 6 * 16 * 80 + (g * 16 + n) * 80 + c * 16);
                async_load_b128(la, ga);
            }
        }
    };
    auto stageM = [&](int buf, int k0) {
        #pragma unroll
        for (int it = 0; it < 8; it++) {
            int idx = tid + 256 * it;
            int mr = idx >> 4, p = idx & 15;
            int row = row0 + mr;
            if (row > Mrows - 1) row = Mrows - 1;
            float2 vm = *(const float2*)(Mm + (size_t)row * Hdim + k0 + 2 * p);
            ldsM[buf][mr][p] = pack2(f2bf(vm.x), f2bf(vm.y));
        }
    };

    issue_asyncH(0, 0);
    issue_asyncB(0, 0);
    stageM(0, 0);

    const int lm  = lane & 15;
    const int kb2 = (lane >> 4) * 4;
    const int bb  = (lane >> 4) * 4;
    const int r = wave * 16 + lm;

    for (int s = 0; s < nsteps; s++) {
        wait_async0();
        __syncthreads();
        if (s + 1 < nsteps) {
            issue_asyncH((s + 1) & 1, 32 * (s + 1));
            issue_asyncB((s + 1) & 1, 32 * (s + 1));
            stageM((s + 1) & 1, 32 * (s + 1));
        }
        const int cur = s & 1;
        const unsigned* __restrict__ Hbuf = &ldsH[cur * 128 * ROWU];
        const unsigned* __restrict__ Bbuf = &ldsB[cur * 6 * 16 * ROWU];

        U8 tm, th;
        #pragma unroll
        for (int p = 0; p < 4; p++) {
            tm.u[p] = ldsM[cur][r][kb2 + p]; tm.u[4 + p] = ldsM[cur][r][8 + kb2 + p];
            th.u[p] = Hbuf[r * ROWU + kb2 + p]; th.u[4 + p] = Hbuf[r * ROWU + 8 + kb2 + p];
        }
        v16bf am = frag_cast(tm), ah = frag_cast(th);
        #pragma unroll
        for (int g = 0; g < 3; g++) {
            U8 tb;
            int base = (g * 16 + lm) * ROWU + bb;
            #pragma unroll
            for (int p = 0; p < 8; p++) tb.u[p] = Bbuf[base + (p & 3) + (p >> 2) * 8];
            aI[g] = wmma_bf16(am, frag_cast(tb), aI[g]);
        }
        #pragma unroll
        for (int g = 0; g < 3; g++) {
            U8 tb;
            int base = ((3 + g) * 16 + lm) * ROWU + bb;
            #pragma unroll
            for (int p = 0; p < 8; p++) tb.u[p] = Bbuf[base + (p & 3) + (p >> 2) * 8];
            aH[g] = wmma_bf16(ah, frag_cast(tb), aH[g]);
        }
    }

    // ---- gate epilogue ----
    const int col = col0 + (lane & 15);
    const int rbase = row0 + wave * 16 + (lane >> 4) * 8;
    const float b_ir = bih[col],            b_hr = bhh[col];
    const float b_iz = bih[Hdim + col],     b_hz = bhh[Hdim + col];
    const float b_in = bih[2 * Hdim + col], b_hn = bhh[2 * Hdim + col];
    #pragma unroll
    for (int v = 0; v < 8; v++) {
        int row = rbase + v;
        if (row < Mrows) {
            float rg = sigmoidf_fast((aI[0][v] + b_ir) + (aH[0][v] + b_hr));
            float zg = sigmoidf_fast((aI[1][v] + b_iz) + (aH[1][v] + b_hz));
            float ng = tanhf_fast((aI[2][v] + b_in) + rg * (aH[2][v] + b_hn));
            float hold = Hm[(size_t)row * Hdim + col];
            float hn = (1.f - zg) * ng + zg * hold;
            Hout[(size_t)row * Hdim + col] = hn;
            Hbf[(size_t)row * Hdim + col]  = f2bf(hn);
        }
    }
}

// ---------------------------------------------------------------------------
// Global mean pool + head MLP
// ---------------------------------------------------------------------------
__global__ void zero_kernel(float* p, int n) {
    int i = blockIdx.x * blockDim.x + threadIdx.x;
    if (i < n) p[i] = 0.f;
}

__global__ __launch_bounds__(256) void pool_accum_kernel(
    const float* __restrict__ h, const int* __restrict__ batch,
    float* __restrict__ pooled, float* __restrict__ cnt, int N) {
    int i = blockIdx.x;
    if (i >= N) return;
    int b = batch[i];
    for (int c = threadIdx.x; c < Hdim; c += 256)
        atomicAdd(&pooled[(size_t)b * Hdim + c], h[(size_t)i * Hdim + c]);
    if (threadIdx.x == 0) atomicAdd(&cnt[b], 1.f);
}

__global__ __launch_bounds__(256) void head_kernel(
    const float* __restrict__ pooled, const float* __restrict__ cnt,
    const float* __restrict__ w1, const float* __restrict__ b1,
    const float* __restrict__ w2, const float* __restrict__ b2,
    float* __restrict__ out) {
    int g = blockIdx.x;
    int t = threadIdx.x;
    __shared__ float red[256];
    float inv = 1.f / fmaxf(cnt[g], 1.f);
    float acc = b1[t];
    for (int k = 0; k < Hdim; k++)
        acc = fmaf(pooled[(size_t)g * Hdim + k] * inv, w1[k * 256 + t], acc);
    acc = fmaxf(acc, 0.f);
    red[t] = acc * w2[t];
    __syncthreads();
    for (int s = 128; s > 0; s >>= 1) {
        if (t < s) red[t] += red[t + s];
        __syncthreads();
    }
    if (t == 0) out[g] = red[0] + b2[0];
}

// ---------------------------------------------------------------------------
// Host orchestration
// ---------------------------------------------------------------------------
extern "C" void kernel_launch(void* const* d_in, const int* in_sizes, int n_in,
                              void* d_out, int out_size, void* d_ws, size_t ws_size,
                              hipStream_t stream) {
    const int N = in_sizes[0] / 24;       // 30000
    const int E = in_sizes[1] / 2;        // 480000
    const int G = 64, L = 6;
    const int H = Hdim;

    const float* x       = (const float*)d_in[0];
    const int*   edge    = (const int*)d_in[1];
    const int*   batch   = (const int*)d_in[2];
    const float* enc_w   = (const float*)d_in[3];
    const float* enc_b   = (const float*)d_in[4];
    const float* mlp_w1  = (const float*)d_in[5];
    const float* mlp_b1  = (const float*)d_in[6];
    const float* mlp_w2  = (const float*)d_in[7];
    const float* mlp_b2  = (const float*)d_in[8];
    const float* gru_wih = (const float*)d_in[9];
    const float* gru_whh = (const float*)d_in[10];
    const float* gru_bih = (const float*)d_in[11];
    const float* gru_bhh = (const float*)d_in[12];
    const float* head_w1 = (const float*)d_in[13];
    const float* head_b1 = (const float*)d_in[14];
    const float* head_w2 = (const float*)d_in[15];
    const float* head_b2 = (const float*)d_in[16];

    auto align256 = [](size_t v) { return (v + 255) & ~(size_t)255; };
    char* base = (char*)d_ws;
    size_t off = 0;
    const size_t actB  = (size_t)N * H * sizeof(float);
    const size_t actBh = (size_t)N * H * sizeof(unsigned short);
    const int K2 = H / 2;

    float* h0   = (float*)(base + off); off += align256(actB);
    float* h1   = (float*)(base + off); off += align256(actB);
    float* msg  = (float*)(base + off); off += align256(actB);
    float* mbuf = (float*)(base + off); off += align256(actB);
    unsigned short* hbf   = (unsigned short*)(base + off); off += align256(actBh);
    unsigned short* tmpbf = (unsigned short*)(base + off); off += align256(actBh);
    unsigned* w1t  = (unsigned*)(base + off); off += align256((size_t)L * H * K2 * 4);
    unsigned* w2t  = (unsigned*)(base + off); off += align256((size_t)L * H * K2 * 4);
    unsigned* wiht = (unsigned*)(base + off); off += align256((size_t)L * 3 * H * K2 * 4);
    unsigned* whht = (unsigned*)(base + off); off += align256((size_t)L * 3 * H * K2 * 4);
    float* pooled = (float*)(base + off); off += align256((size_t)G * H * sizeof(float));
    float* cnt    = (float*)(base + off); off += align256((size_t)G * sizeof(float));

    // 1) weight conversion to pair-packed column-major bf16
    convert_transpose_kernel<<<2048, 256, 0, stream>>>(mlp_w1,  w1t,  H, H,
                                                       (long)L * H * K2);
    convert_transpose_kernel<<<2048, 256, 0, stream>>>(mlp_w2,  w2t,  H, H,
                                                       (long)L * H * K2);
    convert_transpose_kernel<<<4096, 256, 0, stream>>>(gru_wih, wiht, H, 3 * H,
                                                       (long)L * 3 * H * K2);
    convert_transpose_kernel<<<4096, 256, 0, stream>>>(gru_whh, whht, H, 3 * H,
                                                       (long)L * 3 * H * K2);

    // 2) node encoder (f32 h + bf16 shadow)
    encoder_kernel<<<N, 256, 0, stream>>>(x, enc_w, enc_b, h0, hbf, N);

    // 3) message-passing layers
    float* hcur = h0;
    float* tmp  = h1;
    const int mb = (N + 127) / 128;
    for (int l = 0; l < L; l++) {
        // msg MLP: tmp_bf = relu(h @ w1 + b1)   [bf16 out]
        gemm_bias_kernel<<<dim3(mb, H / 64), 256, 0, stream>>>(
            hbf, w1t + (size_t)l * H * K2, mlp_b1 + (size_t)l * H,
            nullptr, tmpbf, N, H, H, 1);
        // msg = tmp_bf @ w2 + b2   [f32 out for scatter atomics]
        gemm_bias_kernel<<<dim3(mb, H / 64), 256, 0, stream>>>(
            tmpbf, w2t + (size_t)l * H * K2, mlp_b2 + (size_t)l * H,
            msg, nullptr, N, H, H, 0);
        copy4_kernel<<<2048, 256, 0, stream>>>((const float4*)msg, (float4*)mbuf,
                                               (long)N * H / 4);
        scatter_kernel<<<E, 128, 0, stream>>>(edge, msg, mbuf, E);
        gru_kernel<<<dim3(mb, H / 16), 256, 0, stream>>>(
            mbuf, hcur, hbf,
            wiht + (size_t)l * 3 * H * K2, whht + (size_t)l * 3 * H * K2,
            gru_bih + (size_t)l * 3 * H, gru_bhh + (size_t)l * 3 * H, tmp, hbf, N);
        float* t2 = hcur; hcur = tmp; tmp = t2;
    }

    // 4) global mean pool + head
    zero_kernel<<<(G * H + 255) / 256, 256, 0, stream>>>(pooled, G * H);
    zero_kernel<<<1, 256, 0, stream>>>(cnt, G);
    pool_accum_kernel<<<N, 256, 0, stream>>>(hcur, batch, pooled, cnt, N);
    head_kernel<<<G, 256, 0, stream>>>(pooled, cnt, head_w1, head_b1, head_w2, head_b2,
                                       (float*)d_out);
}
